// _AttentionInput_8572754723487
// MI455X (gfx1250) — compile-verified, matrix operand order checked by reading, not run on hardware
//
#include <hip/hip_runtime.h>
#include <hip/hip_bf16.h>
#include <math.h>

typedef __attribute__((ext_vector_type(16))) _Float16 v16h;
typedef __attribute__((ext_vector_type(8)))  _Float16 v8h;
typedef __attribute__((ext_vector_type(8)))  float    v8f;

#define L_DIM   2048
#define B_DIM   32
#define E_DIM   512
#define H_DIM   512
#define HID_DIM 512
#define A_DIM   512
#define NEG_INF_F (-10000000000.0f)

// ---------------------------------------------------------------------------
// Prep 1: pre0[b,h] = hidden[b,:] @ W1[:H,:][:,h] + b1[h]   (32 x 512, tiny)
// ---------------------------------------------------------------------------
__global__ void k_pre0(const float* __restrict__ hidden,
                       const float* __restrict__ W1,
                       const float* __restrict__ b1,
                       float* __restrict__ pre0) {
  __shared__ float sh[H_DIM];
  const int b = blockIdx.x;
  for (int k = threadIdx.x; k < H_DIM; k += blockDim.x)
    sh[k] = hidden[b * H_DIM + k];
  __syncthreads();
  for (int h = threadIdx.x; h < HID_DIM; h += blockDim.x) {
    float acc = b1[h];
    #pragma unroll 8
    for (int k = 0; k < H_DIM; ++k)
      acc += sh[k] * W1[k * HID_DIM + h];
    pre0[b * HID_DIM + h] = acc;
  }
}

// ---------------------------------------------------------------------------
// Prep 2: pack W1_e (rows H..H+E-1 of W1) to f16 in WMMA B-operand layout:
//   packW[nt][kc][lane][i], i=0..15 contiguous per lane.
//   lane<16 : holds column n = lane,      K = kc*32 + {0..7, 16..23}
//   lane>=16: holds column n = lane-16,   K = kc*32 + {8..15, 24..31}
// ---------------------------------------------------------------------------
__global__ void k_packW(const float* __restrict__ W1,
                        _Float16* __restrict__ packW) {
  const int idx  = blockIdx.x * blockDim.x + threadIdx.x;  // 0..262143
  const int i    = idx & 15;
  const int lane = (idx >> 4) & 31;
  const int kc   = (idx >> 9) & 15;
  const int nt   = idx >> 13;
  const int n    = nt * 16 + (lane & 15);
  const int kb   = kc * 32 + ((lane >> 4) << 3);
  const int k    = kb + (i < 8 ? i : 8 + i);
  packW[idx] = (_Float16)W1[(H_DIM + k) * HID_DIM + n];
}

// ---------------------------------------------------------------------------
// Pass 1: logit[m] for m = l*32+b. One wave per 16-row M-tile.
//   A-slab (16 rows x 512) converted f32->f16 into LDS in A-operand layout,
//   then 32 N-tiles x 16 K-steps of v_wmma_f32_16x16x32_f16, fused
//   tanh() @ W2 reduction, mask, bias.
// ---------------------------------------------------------------------------
__global__ void __launch_bounds__(128)
k_logits(const float* __restrict__ x,
         const int*   __restrict__ mask,
         const float* __restrict__ pre0,
         const _Float16* __restrict__ packW,
         const float* __restrict__ W2,
         const float* __restrict__ b2,
         float* __restrict__ logit) {
  __shared__ __align__(32) _Float16 sA[4][16 * E_DIM];  // 64 KB

  const int lane = threadIdx.x & 31;
  const int wv   = threadIdx.x >> 5;
  const int tile = blockIdx.x * 4 + wv;        // 0..4095
  const int row  = lane & 15;
  const int hv   = lane >> 4;                  // half-wave id
  const long m0  = (long)tile * 16;

  // ---- load + convert the wave's 16x512 A slab into LDS (A-operand layout)
  const float* xrow = x + (m0 + row) * E_DIM;
  #pragma unroll 4
  for (int kc = 0; kc < 16; ++kc) {
    const int base = kc * 32 + hv * 8;
    float4 f0 = *(const float4*)(xrow + base);
    float4 f1 = *(const float4*)(xrow + base + 4);
    float4 f2 = *(const float4*)(xrow + base + 16);
    float4 f3 = *(const float4*)(xrow + base + 20);
    v8h h0, h1;
    h0[0]=(_Float16)f0.x; h0[1]=(_Float16)f0.y; h0[2]=(_Float16)f0.z; h0[3]=(_Float16)f0.w;
    h0[4]=(_Float16)f1.x; h0[5]=(_Float16)f1.y; h0[6]=(_Float16)f1.z; h0[7]=(_Float16)f1.w;
    h1[0]=(_Float16)f2.x; h1[1]=(_Float16)f2.y; h1[2]=(_Float16)f2.z; h1[3]=(_Float16)f2.w;
    h1[4]=(_Float16)f3.x; h1[5]=(_Float16)f3.y; h1[6]=(_Float16)f3.z; h1[7]=(_Float16)f3.w;
    *(v8h*)&sA[wv][kc * E_DIM + lane * 16]     = h0;
    *(v8h*)&sA[wv][kc * E_DIM + lane * 16 + 8] = h1;
  }
  __syncthreads();

  float accL[8];
  #pragma unroll
  for (int j = 0; j < 8; ++j) accL[j] = 0.0f;

  const int bbase = (tile & 1) << 4;           // b = bbase + tile-row

  for (int nt = 0; nt < 32; ++nt) {
    v8f acc = {};
    const _Float16* bp = packW + nt * (16 * 32 * 16) + lane * 16;
    #pragma unroll
    for (int kc = 0; kc < 16; ++kc) {
      v16h a = *(const v16h*)&sA[wv][kc * E_DIM + lane * 16];
      v16h b = *(const v16h*)(bp + kc * (32 * 16));
      acc = __builtin_amdgcn_wmma_f32_16x16x32_f16(
          false, a, false, b, (short)0, acc, false, false);
    }
    // C-tile: VGPR j -> row (j + 8*hv), column nt*16 + (lane&15)
    const int   hcol = nt * 16 + (lane & 15);
    const float w2v  = W2[hcol];
    #pragma unroll
    for (int j = 0; j < 8; ++j) {
      const int r  = j + hv * 8;
      const int bb = bbase + r;
      float pre = acc[j] + pre0[bb * HID_DIM + hcol];
      accL[j] += tanhf(pre) * w2v;
    }
  }

  // reduce over the 16 N-lanes of each half-wave (xor 1,2,4,8 stays in half)
  #pragma unroll
  for (int j = 0; j < 8; ++j) {
    float v = accL[j];
    v += __shfl_xor(v, 1, 32);
    v += __shfl_xor(v, 2, 32);
    v += __shfl_xor(v, 4, 32);
    v += __shfl_xor(v, 8, 32);
    accL[j] = v;
  }

  if ((lane & 15) == 0) {
    const float bias = b2[0];
    #pragma unroll
    for (int j = 0; j < 8; ++j) {
      const long m = m0 + j + hv * 8;
      float lg = accL[j] + bias;
      if (mask[m] == 0) lg = NEG_INF_F;
      logit[m] = lg;
    }
  }
}

// ---------------------------------------------------------------------------
// Pass 2: column softmax over L (axis 0). One block per b.
// ---------------------------------------------------------------------------
__global__ void k_softmax(const float* __restrict__ logit,
                          float* __restrict__ p) {
  __shared__ float red[256];
  const int b   = blockIdx.x;
  const int tid = threadIdx.x;

  float mx = -INFINITY;
  for (int l = tid; l < L_DIM; l += 256)
    mx = fmaxf(mx, logit[l * B_DIM + b]);
  red[tid] = mx; __syncthreads();
  for (int s = 128; s > 0; s >>= 1) {
    if (tid < s) red[tid] = fmaxf(red[tid], red[tid + s]);
    __syncthreads();
  }
  mx = red[0]; __syncthreads();

  float sum = 0.0f;
  for (int l = tid; l < L_DIM; l += 256)
    sum += __expf(logit[l * B_DIM + b] - mx);
  red[tid] = sum; __syncthreads();
  for (int s = 128; s > 0; s >>= 1) {
    if (tid < s) red[tid] += red[tid + s];
    __syncthreads();
  }
  const float inv = 1.0f / red[0];

  for (int l = tid; l < L_DIM; l += 256)
    p[l * B_DIM + b] = __expf(logit[l * B_DIM + b] - mx) * inv;
}

// ---------------------------------------------------------------------------
// Pass 3: ctx[b,e] = sum_l p[l,b] * x[l,b,e]; fused concat of action_feature.
// grid: 64 blocks (2 per b), 256 threads; p staged in LDS.
// ---------------------------------------------------------------------------
__global__ void k_ctx(const float* __restrict__ x,
                      const float* __restrict__ p,
                      const float* __restrict__ act,
                      float* __restrict__ out) {
  __shared__ float sP[256];
  const int b = blockIdx.x >> 1;
  const int e = ((blockIdx.x & 1) << 8) + threadIdx.x;

  float acc = 0.0f;
  for (int l0 = 0; l0 < L_DIM; l0 += 256) {
    sP[threadIdx.x] = p[(l0 + threadIdx.x) * B_DIM + b];
    __syncthreads();
    #pragma unroll 8
    for (int i = 0; i < 256; ++i)
      acc += x[((long)(l0 + i) * B_DIM + b) * E_DIM + e] * sP[i];
    __syncthreads();
  }
  out[b * (E_DIM + A_DIM) + e]         = acc;
  out[b * (E_DIM + A_DIM) + E_DIM + e] = act[b * A_DIM + e];
}

// ---------------------------------------------------------------------------
extern "C" void kernel_launch(void* const* d_in, const int* in_sizes, int n_in,
                              void* d_out, int out_size, void* d_ws, size_t ws_size,
                              hipStream_t stream) {
  const float* x    = (const float*)d_in[0];   // (L,B,E)
  const int*   mask = (const int*)  d_in[1];   // (L,B)
  const float* act  = (const float*)d_in[2];   // (B,A)
  const float* hid  = (const float*)d_in[3];   // (B,H)
  // d_in[4] = state, unused by the reference
  const float* W1   = (const float*)d_in[5];   // (H+E, HID)
  const float* b1   = (const float*)d_in[6];   // (HID,)
  const float* W2   = (const float*)d_in[7];   // (HID,1)
  const float* b2   = (const float*)d_in[8];   // (1,)
  float* out = (float*)d_out;

  char* ws = (char*)d_ws;
  float*    pre0  = (float*)(ws);                      //  64 KB
  float*    logit = (float*)(ws + (64u  << 10));       // 256 KB
  float*    p     = (float*)(ws + (320u << 10));       // 256 KB
  _Float16* packW = (_Float16*)(ws + (576u << 10));    // 512 KB

  k_pre0   <<<B_DIM, 256, 0, stream>>>(hid, W1, b1, pre0);
  k_packW  <<<1024,  256, 0, stream>>>(W1, packW);
  k_logits <<<1024,  128, 0, stream>>>(x, mask, pre0, packW, W2, b2, logit);
  k_softmax<<<B_DIM, 256, 0, stream>>>(logit, p);
  k_ctx    <<<64,    256, 0, stream>>>(x, p, act, out);
}
